// VectorQuantizer_43765716746431
// MI455X (gfx1250) — matrix operand order repaired and checked by reading, not compiled, not run on hardware
//
#include <hip/hip_runtime.h>
#include <hip/hip_bf16.h>
#include <math.h>
#include <stdint.h>

typedef __bf16 bf16_t;
typedef __bf16 v16bf __attribute__((ext_vector_type(16)));
typedef __bf16 v8bf  __attribute__((ext_vector_type(8)));
typedef float  v8f   __attribute__((ext_vector_type(8)));

#define NUM_CODES 8192
#define CODE_DIM  256
#define BN        16384        // B*N = 4*4096
#define PROW      40           // LDS row stride (elements) for P tile: 80B, 16B-aligned
#define LOG2E     1.4426950408889634f
#define NLOG2LN2  0.5287663729448977f   // -log2(ln 2)

// ---- gfx1250 async global->LDS copy (ASYNCcnt-tracked), GVS addressing ----
__device__ __forceinline__ void async_ld16(uint32_t lds_off, uint32_t goff, uint64_t sbase) {
    asm volatile("global_load_async_to_lds_b128 %0, %1, %2 offset:0"
                 :: "v"(lds_off), "v"(goff), "s"(sbase) : "memory");
}
__device__ __forceinline__ void wait_async0() {
    asm volatile("s_wait_asynccnt 0" ::: "memory");
}

// ---------------- pass 1a: z -> bf16 scaled by 2*log2(e) (folds GEMM scale + base-2 softmax) ----
__global__ void vq_convert_z(const float* __restrict__ z, bf16_t* __restrict__ zbf, int n) {
    int i = blockIdx.x * blockDim.x + threadIdx.x;
    if (i < n) zbf[i] = (bf16_t)(2.0f * LOG2E * z[i]);
}

// ---------------- pass 1b: codebook -> bf16 (row major + transposed) and log2(e)*||c||^2 ----------
__global__ void vq_prep_codebook(const float* __restrict__ cb,
                                 bf16_t* __restrict__ cbbf,
                                 bf16_t* __restrict__ cbT,
                                 float*  __restrict__ csq) {
    __shared__ float red[CODE_DIM];
    int c = blockIdx.x;      // code index
    int d = threadIdx.x;     // dim index
    float v = cb[(size_t)c * CODE_DIM + d];
    cbbf[(size_t)c * CODE_DIM + d]   = (bf16_t)v;
    cbT [(size_t)d * NUM_CODES + c]  = (bf16_t)v;
    red[d] = v * v;
    __syncthreads();
    for (int s = CODE_DIM / 2; s > 0; s >>= 1) {
        if (d < s) red[d] += red[d + s];
        __syncthreads();
    }
    if (d == 0) csq[c] = LOG2E * red[0];
}

// ---------------- pass 2: fused flash-style gumbel-softmax VQ (base-2 domain) ----------------
// 8 waves / block, each wave32 owns 16 token rows; codebook chunk async-staged once per block.
__global__ __launch_bounds__(256, 1)
void vq_fused(const float* __restrict__ z,
              const float* __restrict__ u_noise,
              const bf16_t* __restrict__ zbf,
              const bf16_t* __restrict__ cbbf,
              const bf16_t* __restrict__ cbT,
              const float* __restrict__ csq,
              float* __restrict__ out,
              float* __restrict__ loss) {
    __shared__ __align__(16) bf16_t cbA_s[32 * CODE_DIM];   // chunk, code-major  (16KB)
    __shared__ __align__(16) bf16_t cbB_s[CODE_DIM * 32];   // chunk, dim-major   (16KB)
    __shared__ __align__(16) bf16_t pbuf[8][16 * PROW];     // per-wave P bounce  (10KB)

    const int tid  = threadIdx.x;
    const int wave = tid >> 5;
    const int lane = tid & 31;
    const int h    = lane >> 4;   // half-of-wave (WMMA layout halves)
    const int ln   = lane & 15;
    const int rowbase = (blockIdx.x * 8 + wave) * 16;

    bf16_t* P = pbuf[wave];
    const uint32_t ldsA   = (uint32_t)(uintptr_t)cbA_s;
    const uint32_t ldsB   = (uint32_t)(uintptr_t)cbB_s;
    const uint64_t gBbase = (uint64_t)(uintptr_t)cbT;

    v8f o[16] = {};               // z_q accumulator: 16 dim-tiles x (16x16 f32)
    float m[8], l[8];
    #pragma unroll
    for (int r = 0; r < 8; ++r) { m[r] = -3.0e38f; l[r] = 0.0f; }

    for (int cc = 0; cc < NUM_CODES; cc += 32) {
        // ---- cooperative async stage of the 32-code chunk into LDS (both layouts) ----
        {
            const uint64_t gA = (uint64_t)(uintptr_t)(cbbf + (size_t)cc * CODE_DIM);
            #pragma unroll
            for (int j = 0; j < 4; ++j) {
                int seg = j * 256 + tid;            // 1024 x 16B segments per buffer
                async_ld16(ldsA + seg * 16, (uint32_t)(seg * 16), gA);
                int row = seg >> 2, c4 = seg & 3;
                async_ld16(ldsB + seg * 16,
                           (uint32_t)(row * NUM_CODES * 2 + cc * 2 + c4 * 16), gBbase);
            }
            wait_async0();
        }
        __syncthreads();

        if (cc + 32 < NUM_CODES)
            __builtin_prefetch(&u_noise[(size_t)rowbase * NUM_CODES + cc + 32], 0, 1);

        // ---- scores (log2 domain): S' = (2*log2e*Z)@C^T + (g' - log2e*||c||^2) ----
        // g' = -log2(-log2(u)) - log2(ln2): two raw v_log_f32, no OCML guards.
        v8f s0, s1;
        {
            float c0 = csq[cc + ln];
            float c1 = csq[cc + 16 + ln];
            #pragma unroll
            for (int r = 0; r < 8; ++r) {
                size_t urow = (size_t)(rowbase + r + 8 * h) * NUM_CODES + cc;
                float u0 = u_noise[urow + ln];
                float u1 = u_noise[urow + 16 + ln];
                float t0 = __builtin_amdgcn_logf(-__builtin_amdgcn_logf(u0));
                float t1 = __builtin_amdgcn_logf(-__builtin_amdgcn_logf(u1));
                s0[r] = (NLOG2LN2 - t0) - c0;
                s1[r] = (NLOG2LN2 - t1) - c1;
            }
        }
        #pragma unroll
        for (int d = 0; d < CODE_DIM; d += 32) {
            v16bf a;
            {   // A-layout: row = ln, 16 contiguous K at 16h
                const bf16_t* zr = zbf + (size_t)(rowbase + ln) * CODE_DIM + d;
                v8bf lo = *(const v8bf*)(zr + 16 * h);
                v8bf hi = *(const v8bf*)(zr + 16 * h + 8);
                #pragma unroll
                for (int i = 0; i < 8; ++i) { a[i] = lo[i]; a[i + 8] = hi[i]; }
            }
            // B-layout from LDS: col = code = ln, 16 contiguous dims per lane at 16h
            v16bf b0 = *(const v16bf*)(cbA_s + (ln)      * CODE_DIM + d + 16 * h);
            v16bf b1 = *(const v16bf*)(cbA_s + (16 + ln) * CODE_DIM + d + 16 * h);
            s0 = __builtin_amdgcn_wmma_f32_16x16x32_bf16(false, a, false, b0, (short)0, s0, false, false);
            s1 = __builtin_amdgcn_wmma_f32_16x16x32_bf16(false, a, false, b1, (short)0, s1, false, false);
        }

        // ---- online softmax in base 2 (rows live across 16-lane halves) ----
        float scale[8];
        #pragma unroll
        for (int r = 0; r < 8; ++r) {
            float mc = fmaxf(s0[r], s1[r]);
            #pragma unroll
            for (int k = 1; k < 16; k <<= 1) mc = fmaxf(mc, __shfl_xor(mc, k, 32));
            float mn = fmaxf(m[r], mc);
            scale[r] = __builtin_amdgcn_exp2f(m[r] - mn);
            float p0 = __builtin_amdgcn_exp2f(s0[r] - mn);
            float p1 = __builtin_amdgcn_exp2f(s1[r] - mn);
            s0[r] = p0; s1[r] = p1;
            float rs = p0 + p1;
            #pragma unroll
            for (int k = 1; k < 16; k <<= 1) rs += __shfl_xor(rs, k, 32);
            l[r] = l[r] * scale[r] + rs;
            m[r] = mn;
        }
        #pragma unroll
        for (int t = 0; t < 16; ++t) {
            #pragma unroll
            for (int r = 0; r < 8; ++r) o[t][r] *= scale[r];
        }

        // ---- C-layout -> A-layout via per-wave LDS bounce (bf16) ----
        #pragma unroll
        for (int r = 0; r < 8; ++r) {
            P[(r + 8 * h) * PROW + ln]      = (bf16_t)s0[r];
            P[(r + 8 * h) * PROW + 16 + ln] = (bf16_t)s1[r];
        }
        v16bf pa;
        {
            const bf16_t* pr = P + ln * PROW;     // per-wave buffer; wave-internal DS ordering
            v8bf lo = *(const v8bf*)(pr + 16 * h);
            v8bf hi = *(const v8bf*)(pr + 16 * h + 8);
            #pragma unroll
            for (int i = 0; i < 8; ++i) { pa[i] = lo[i]; pa[i + 8] = hi[i]; }
        }

        // ---- O += P @ C_chunk (codes are K, dims are N) over 16 dim-tiles ----
        #pragma unroll
        for (int t = 0; t < 16; ++t) {
            v16bf b = *(const v16bf*)(cbB_s + (t * 16 + ln) * 32 + 16 * h);
            o[t] = __builtin_amdgcn_wmma_f32_16x16x32_bf16(false, pa, false, b, (short)0, o[t], false, false);
        }
        __syncthreads();   // all waves done with staged chunk before restage
    }

    // ---- finalize: z_q = O / l ; loss partial = sum (z_q - z)^2 ----
    float lacc = 0.0f;
    #pragma unroll
    for (int r = 0; r < 8; ++r) {
        float invl = 1.0f / l[r];
        int row = rowbase + r + 8 * h;
        #pragma unroll
        for (int t = 0; t < 16; ++t) {
            int dim = t * 16 + ln;
            float zq = o[t][r] * invl;
            out[(size_t)row * CODE_DIM + dim] = zq;
            float diff = zq - z[(size_t)row * CODE_DIM + dim];
            lacc += diff * diff;
        }
    }
    #pragma unroll
    for (int k = 1; k < 32; k <<= 1) lacc += __shfl_xor(lacc, k, 32);
    if (lane == 0)
        atomicAdd(loss, lacc * (2.0f / ((float)BN * (float)CODE_DIM)));
}

extern "C" void kernel_launch(void* const* d_in, const int* in_sizes, int n_in,
                              void* d_out, int out_size, void* d_ws, size_t ws_size,
                              hipStream_t stream) {
    const float* z  = (const float*)d_in[0];   // [4,4096,256]
    const float* cb = (const float*)d_in[1];   // [8192,256]
    const float* u  = (const float*)d_in[2];   // [16384,8192]

    char* ws = (char*)d_ws;
    bf16_t* zbf  = (bf16_t*)(ws);                              // 8,388,608 B
    bf16_t* cbbf = (bf16_t*)(ws + 8388608);                    // 4,194,304 B
    bf16_t* cbT  = (bf16_t*)(ws + 8388608 + 4194304);          // 4,194,304 B
    float*  csq  = (float* )(ws + 8388608 + 2 * 4194304);      //    32,768 B

    float* out  = (float*)d_out;
    float* loss = out + (out_size - 1);

    (void)hipMemsetAsync(loss, 0, sizeof(float), stream);
    vq_convert_z<<<(BN * CODE_DIM + 255) / 256, 256, 0, stream>>>(z, zbf, BN * CODE_DIM);
    vq_prep_codebook<<<NUM_CODES, CODE_DIM, 0, stream>>>(cb, cbbf, cbT, csq);
    vq_fused<<<BN / 128, 256, 0, stream>>>(z, u, zbf, cbbf, cbT, csq, out, loss);
}
